// WeightedCorrelationLayer_10488310136851
// MI455X (gfx1250) — matrix operand order repaired and checked by reading, not compiled, not run on hardware
//
#include <hip/hip_runtime.h>

typedef _Float16 f16;
typedef __attribute__((ext_vector_type(8)))  _Float16 v8h;
typedef __attribute__((ext_vector_type(16))) _Float16 v16h;
typedef __attribute__((ext_vector_type(8)))  float    v8f;

// Problem constants (from reference)
#define BB   8
#define CC   128
#define LL   16
#define HH   56
#define WW   56
#define KK   7
#define NO   49          // 7*7 offsets
#define GG   4
#define HW   (HH*WW)     // 3136

// LDS layout constants
#define LPAD 136         // padded channel stride (f16): 272B -> conflict-free b128 fetches
#define WDIM 64          // padded row width (w'' = w + dx <= 63)
#define NROW 7

#define XS_ELEMS  (NROW*WDIM*LPAD)
#define X1_ELEMS  (WDIM*LPAD)
#define WT_ELEMS  (NO*LPAD)
#define ZB_ELEMS  32                  // 64-B zero block for non-weight lanes
#define SMEM_ELEMS (XS_ELEMS + X1_ELEMS + WT_ELEMS + ZB_ELEMS)

static __device__ __forceinline__ v16h cat16(v8h lo, v8h hi) {
  return __builtin_shufflevector(lo, hi, 0,1,2,3,4,5,6,7,8,9,10,11,12,13,14,15);
}

__global__ __launch_bounds__(128)
void wcl_wmma_kernel(const float* __restrict__ x,
                     const float* __restrict__ fw,
                     float* __restrict__ out) {
  extern __shared__ f16 smem[];
  f16* xsl = smem;                  // [7][WDIM][LPAD]
  f16* x1l = smem + XS_ELEMS;       // [WDIM][LPAD]
  f16* wtl = x1l + X1_ELEMS;        // [49][LPAD]
  f16* zbl = wtl + WT_ELEMS;        // [32] zeros

  const int h  = blockIdx.x;
  const int t  = blockIdx.y;
  const int b  = blockIdx.z;
  const int tid = threadIdx.x;      // 0..127
  const int t1 = (t == 0) ? 0 : t - 1;

  // ---- Stage shifted-source rows (time t, rows h-3..h+3, cols -3..60), zero-padded, f16
  for (int f = tid; f < NROW*WDIM*CC; f += 128) {
    const int wpp = f & 63;
    const int rc  = f >> 6;
    const int row = rc % 7;
    const int c   = rc / 7;
    const int hr  = h + row - 3;
    const int wp  = wpp - 3;
    float v = 0.0f;
    if ((unsigned)hr < (unsigned)HH && (unsigned)wp < (unsigned)WW)
      v = x[(((size_t)(b*CC + c)*LL + t)*HH + hr)*WW + wp];
    xsl[(row*WDIM + wpp)*LPAD + c] = (f16)v;
  }
  // ---- Stage temporal-shift row x1 = x[t-1 (clamped)] at row h
  for (int f = tid; f < WDIM*CC; f += 128) {
    const int wpp = f & 63;
    const int c   = f >> 6;
    float v = 0.0f;
    if (wpp < WW)
      v = x[(((size_t)(b*CC + c)*LL + t1)*HH + h)*WW + wpp];
    x1l[wpp*LPAD + c] = (f16)v;
  }
  // ---- Stage weights for this t, pre-scaled by 1/32 (the group mean)
  for (int f = tid; f < NO*CC; f += 128) {
    const int c = f & 127;
    const int o = f >> 7;
    const float v = fw[((size_t)c*LL + t)*NO + o];
    wtl[o*LPAD + c] = (f16)(v * (1.0f/32.0f));
  }
  if (tid < ZB_ELEMS) zbl[tid] = (f16)0.0f;
  __syncthreads();

  // ---- Per-wave WMMA compute: wave covers 14 output columns
  const int wid  = tid >> 5;
  const int lane = tid & 31;
  const int n    = lane & 15;       // N column (pixel) / M row
  const int half = lane >> 4;       // K sub-range select
  const int w0   = wid * 14;
  const int w    = w0 + n;          // output column (valid when n < 14)

  // Preload x1 in B layout: chunk j, lane half h -> channels 32j+16h .. +15
  v16h a1[4];
#pragma unroll
  for (int j = 0; j < 4; ++j) {
    const f16* p = &x1l[w*LPAD + 32*j + 16*half];
    a1[j] = cat16(*(const v8h*)p, *(const v8h*)(p + 8));
  }

  // Branch-free shared A operand: lane n<4 carries row n (weights of chunk n),
  // all other lanes read the zero block.  One v_cndmask'd pointer, no exec ops.
  const bool wlane = (n < 4);
  const f16* aBase = wlane ? (wtl + 32*n + 16*half) : zbl;
  const int  aStep = wlane ? LPAD : 0;      // advance per offset o

  const bool dostore = (half == 0) && (n < 14);
  const size_t base00 = (((size_t)b*(NO*GG)*LL + t)*HH + h)*WW + w;

  for (int dy = 0; dy < KK; ++dy) {
    const f16* xsrow = &xsl[dy*WDIM*LPAD];
    for (int dx = 0; dx < KK; ++dx) {
      const int o = dy*KK + dx;

      // Shared A for all 4 chunks of this offset
      const f16* pa = aBase + aStep * o;
      const v16h av = cat16(*(const v8h*)pa, *(const v8h*)(pa + 8));

      const f16* pb0 = &xsrow[(w + dx)*LPAD + 16*half];

      // 4 independent accumulator chains (one per chunk/group); cross-chunk
      // garbage lands in rows we never read.
      v8f acc0 = {}, acc1 = {}, acc2 = {}, acc3 = {};
      {
        v16h bv = cat16(*(const v8h*)(pb0 +  0), *(const v8h*)(pb0 +  8));
        acc0 = __builtin_amdgcn_wmma_f32_16x16x32_f16(false, av, false, a1[0]*bv, (short)0, acc0, false, false);
      }
      {
        v16h bv = cat16(*(const v8h*)(pb0 + 32), *(const v8h*)(pb0 + 40));
        acc1 = __builtin_amdgcn_wmma_f32_16x16x32_f16(false, av, false, a1[1]*bv, (short)0, acc1, false, false);
      }
      {
        v16h bv = cat16(*(const v8h*)(pb0 + 64), *(const v8h*)(pb0 + 72));
        acc2 = __builtin_amdgcn_wmma_f32_16x16x32_f16(false, av, false, a1[2]*bv, (short)0, acc2, false, false);
      }
      {
        v16h bv = cat16(*(const v8h*)(pb0 + 96), *(const v8h*)(pb0 + 104));
        acc3 = __builtin_amdgcn_wmma_f32_16x16x32_f16(false, av, false, a1[3]*bv, (short)0, acc3, false, false);
      }

      if (dostore) {
        float* op = out + base00 + (size_t)o * (GG*LL*HW);
        op[0*LL*HW] = acc0[0];     // group g result = row g of chunk-g accumulator
        op[1*LL*HW] = acc1[1];
        op[2*LL*HW] = acc2[2];
        op[3*LL*HW] = acc3[3];
      }
    }
  }
}

extern "C" void kernel_launch(void* const* d_in, const int* in_sizes, int n_in,
                              void* d_out, int out_size, void* d_ws, size_t ws_size,
                              hipStream_t stream) {
  const float* x  = (const float*)d_in[0];   // (8,128,16,56,56) f32
  const float* fw = (const float*)d_in[1];   // (128,16,7,7) f32
  float* out = (float*)d_out;                // (8,196,16,56,56) f32

  dim3 grid(HH, LL, BB);                     // one workgroup per (h, t, b)
  dim3 block(128);                           // 4 waves, 14 pixels each
  const size_t smem = (size_t)SMEM_ELEMS * sizeof(f16);  // ~149 KB of the 320 KB WGP LDS
  wcl_wmma_kernel<<<grid, block, smem, stream>>>(x, fw, out);
}
